// HRALinear_60266981098045
// MI455X (gfx1250) — compile-verified, matrix operand order checked by reading, not run on hardware
//
#include <hip/hip_runtime.h>
#include <hip/hip_bf16.h>

typedef __attribute__((ext_vector_type(16))) _Float16 v16h;
typedef __attribute__((ext_vector_type(8)))  _Float16 v8h;
typedef __attribute__((ext_vector_type(4)))  _Float16 v4h;
typedef __attribute__((ext_vector_type(8)))  float    v8f;

union Frag { v16h v; v8h h[2]; };

// Problem constants (fixed by the reference)
constexpr int BQ   = 4;
constexpr int SQ   = 2048;
constexpr int DIN  = 4096;   // K
constexpr int DOUT = 4096;   // N
constexpr int RNK  = 8;
constexpr int M    = BQ * SQ;   // 8192
constexpr int N    = DOUT;      // 4096
constexpr int K    = DIN;       // 4096

// GEMM tiling
constexpr int MT  = 128;
constexpr int NT  = 128;
constexpr int KT  = 64;
constexpr int LDT = KT + 8;     // 72 halfwords -> 144B row stride (16B aligned)
constexpr int NIT = K / KT;     // 64 K-slabs

// ---------------------------------------------------------------------------
// 1) inverse norms of the 8 hra_u columns
// ---------------------------------------------------------------------------
__global__ __launch_bounds__(256) void norm_kernel(const float* __restrict__ hra_u,
                                                   float* __restrict__ invn) {
    __shared__ float red[256];
    const int i = blockIdx.x;     // column 0..7
    const int t = threadIdx.x;
    float s = 0.0f;
    for (int d = t; d < DIN; d += 256) {
        float v = hra_u[(size_t)d * RNK + i];
        s += v * v;
    }
    red[t] = s;
    __syncthreads();
    for (int off = 128; off > 0; off >>= 1) {
        if (t < off) red[t] += red[t + off];
        __syncthreads();
    }
    if (t == 0) invn[i] = rsqrtf(red[0]);
}

// ---------------------------------------------------------------------------
// 2) apply 8 sequential Householder reflections per W row, emit f16 W'
//    w <- w - 2*(w . u_i) * u_i  with u_i = hra_u[:,i] * invn[i]
// ---------------------------------------------------------------------------
__global__ __launch_bounds__(256) void update_w_kernel(const float* __restrict__ W,
                                                       const float* __restrict__ hra_u,
                                                       const float* __restrict__ invn,
                                                       _Float16* __restrict__ W16) {
    __shared__ float red[256];
    const int row = blockIdx.x;
    const int t   = threadIdx.x;

    float w[16];
    const float* wr = W + (size_t)row * DIN;
#pragma unroll
    for (int j = 0; j < 16; ++j) w[j] = wr[t + 256 * j];

    for (int i = 0; i < RNK; ++i) {
        const float s = invn[i];
        float dot = 0.0f;
#pragma unroll
        for (int j = 0; j < 16; ++j)
            dot += w[j] * hra_u[(size_t)(t + 256 * j) * RNK + i];
        red[t] = dot;
        __syncthreads();
        for (int off = 128; off > 0; off >>= 1) {
            if (t < off) red[t] += red[t + off];
            __syncthreads();
        }
        const float c = -2.0f * red[0] * s * s;   // folds both normalizations
        __syncthreads();                          // protect red[] before next reuse
#pragma unroll
        for (int j = 0; j < 16; ++j)
            w[j] += c * hra_u[(size_t)(t + 256 * j) * RNK + i];
    }

    _Float16* o = W16 + (size_t)row * DIN;
#pragma unroll
    for (int j = 0; j < 16; ++j) o[t + 256 * j] = (_Float16)w[j];
}

// ---------------------------------------------------------------------------
// 3) x: f32 -> f16 (one-time, halves GEMM A-side bandwidth)
// ---------------------------------------------------------------------------
__global__ __launch_bounds__(256) void convert_x(const float* __restrict__ x,
                                                 _Float16* __restrict__ x16) {
    const size_t idx = ((size_t)blockIdx.x * 256 + threadIdx.x) * 4;
    float4 v = *(const float4*)(x + idx);
    v4h h;
    h[0] = (_Float16)v.x; h[1] = (_Float16)v.y;
    h[2] = (_Float16)v.z; h[3] = (_Float16)v.w;
    *(v4h*)(x16 + idx) = h;
}

// ---------------------------------------------------------------------------
// 4) GEMM: out[m,n] = sum_k x16[m,k] * W16[n,k] + bias[n]
//    128x128 tile / 256 threads (8 waves, 4x2); K in 64-slabs through
//    double-buffered LDS with register staging (1 barrier per slab).
// ---------------------------------------------------------------------------
__global__ __launch_bounds__(256) void gemm_kernel(const _Float16* __restrict__ A,   // [M][K]
                                                   const _Float16* __restrict__ Bw,  // [N][K]
                                                   const float* __restrict__ bias,
                                                   float* __restrict__ out) {
    __shared__ _Float16 sA[2][MT * LDT];
    __shared__ _Float16 sB[2][NT * LDT];

    const int t    = threadIdx.x;
    const int wave = t >> 5;
    const int lane = t & 31;
    const int wm   = wave >> 1;      // 0..3 -> 32 rows each
    const int wn   = wave & 1;       // 0..1 -> 64 cols each
    const int lhalf = lane >> 4;     // 0/1
    const int lmod  = lane & 15;

    const int m0 = blockIdx.x * MT;
    const int n0 = blockIdx.y * NT;

    // Per-thread staging geometry: 1024 x 16B chunks per tile, 4 chunks/thread
    int rowj[4], kcj[4];
    const _Float16* gA[4];
    const _Float16* gB[4];
    int ldsOff[4];
#pragma unroll
    for (int j = 0; j < 4; ++j) {
        const int c = t + 256 * j;
        rowj[j] = c >> 3;
        kcj[j]  = (c & 7) * 8;
        gA[j] = A  + (size_t)(m0 + rowj[j]) * K + kcj[j];
        gB[j] = Bw + (size_t)(n0 + rowj[j]) * K + kcj[j];
        ldsOff[j] = rowj[j] * LDT + kcj[j];
    }

    v8f zero8;
#pragma unroll
    for (int e = 0; e < 8; ++e) zero8[e] = 0.0f;

    v8f acc[2][4];
#pragma unroll
    for (int i = 0; i < 2; ++i)
#pragma unroll
        for (int j = 0; j < 4; ++j) acc[i][j] = zero8;

    // ---- prologue: stage slab 0 into buffer 0 ----
    uint4 ra[4], rb[4];
#pragma unroll
    for (int j = 0; j < 4; ++j) {
        ra[j] = *(const uint4*)(gA[j]);
        rb[j] = *(const uint4*)(gB[j]);
    }
#pragma unroll
    for (int j = 0; j < 4; ++j) {
        *(uint4*)(sA[0] + ldsOff[j]) = ra[j];
        *(uint4*)(sB[0] + ldsOff[j]) = rb[j];
    }
    __syncthreads();

    for (int it = 0; it < NIT; ++it) {
        const int buf = it & 1;

        // (1) issue global loads for the next slab; latency hides under WMMAs
        if (it + 1 < NIT) {
            const int koff = (it + 1) * KT;
#pragma unroll
            for (int j = 0; j < 4; ++j) {
                ra[j] = *(const uint4*)(gA[j] + koff);
                rb[j] = *(const uint4*)(gB[j] + koff);
            }
        }

        // (2) compute on current buffer
        const _Float16* cA = sA[buf];
        const _Float16* cB = sB[buf];
#pragma unroll
        for (int ks = 0; ks < KT; ks += 32) {
            Frag af[2], bf[4];
#pragma unroll
            for (int i = 0; i < 2; ++i) {
                const _Float16* p = cA + (wm * 32 + i * 16 + lmod) * LDT + ks;
                af[i].h[0] = *(const v8h*)(p + 8 * lhalf);        // K = ks + {0..7|8..15}
                af[i].h[1] = *(const v8h*)(p + 16 + 8 * lhalf);   // K = ks + 16 + ...
            }
#pragma unroll
            for (int j = 0; j < 4; ++j) {
                const _Float16* p = cB + (wn * 64 + j * 16 + lmod) * LDT + ks;
                bf[j].h[0] = *(const v8h*)(p + 8 * lhalf);
                bf[j].h[1] = *(const v8h*)(p + 16 + 8 * lhalf);
            }
#pragma unroll
            for (int i = 0; i < 2; ++i)
#pragma unroll
                for (int j = 0; j < 4; ++j)
                    acc[i][j] = __builtin_amdgcn_wmma_f32_16x16x32_f16(
                        false, af[i].v, false, bf[j].v,
                        (short)0, acc[i][j], false, false);
        }

        // (3) store staged registers into the other buffer (safe: it was last
        //     read before the previous barrier), then (4) one barrier per slab
        if (it + 1 < NIT) {
            _Float16* nA = sA[buf ^ 1];
            _Float16* nB = sB[buf ^ 1];
#pragma unroll
            for (int j = 0; j < 4; ++j) {
                *(uint4*)(nA + ldsOff[j]) = ra[j];
                *(uint4*)(nB + ldsOff[j]) = rb[j];
            }
        }
        __syncthreads();
    }

    // Epilogue: C/D layout -> VGPR v: M = base + v + 8*lhalf, N = base + lmod
#pragma unroll
    for (int i = 0; i < 2; ++i) {
        const int mbase = m0 + wm * 32 + i * 16 + lhalf * 8;
#pragma unroll
        for (int j = 0; j < 4; ++j) {
            const int n  = n0 + wn * 64 + j * 16 + lmod;
            const float bv = bias[n];
#pragma unroll
            for (int v = 0; v < 8; ++v)
                out[(size_t)(mbase + v) * N + n] = acc[i][j][v] + bv;
        }
    }
}

// ---------------------------------------------------------------------------
// launch
// ---------------------------------------------------------------------------
extern "C" void kernel_launch(void* const* d_in, const int* in_sizes, int n_in,
                              void* d_out, int out_size, void* d_ws, size_t ws_size,
                              hipStream_t stream) {
    const float* x      = (const float*)d_in[0];
    const float* hra_u  = (const float*)d_in[1];
    const float* base_w = (const float*)d_in[2];
    const float* bias   = (const float*)d_in[3];
    float*       out    = (float*)d_out;

    char* ws = (char*)d_ws;
    _Float16* W16  = (_Float16*)ws;                                           // 32 MB
    _Float16* X16  = (_Float16*)(ws + (size_t)DOUT * DIN * sizeof(_Float16)); // +64 MB
    float*    invn = (float*)(ws + (size_t)DOUT * DIN * sizeof(_Float16)
                                 + (size_t)M * DIN * sizeof(_Float16));       // 8 floats

    norm_kernel<<<RNK, 256, 0, stream>>>(hra_u, invn);
    update_w_kernel<<<DOUT, 256, 0, stream>>>(base_w, hra_u, invn, W16);
    convert_x<<<(M * (size_t)K) / (4 * 256), 256, 0, stream>>>(x, X16);

    dim3 grid(M / MT, N / NT);
    gemm_kernel<<<grid, 256, 0, stream>>>(X16, W16, bias, out);
}
